// HiPPO_LegT_7516192768050
// MI455X (gfx1250) — matrix operand order repaired and checked by reading, not compiled, not run on hardware
//
#include <hip/hip_runtime.h>
#include <hip/hip_bf16.h>

typedef __attribute__((ext_vector_type(16))) _Float16 v16h;
typedef __attribute__((ext_vector_type(8)))  _Float16 v8h;
typedef __attribute__((ext_vector_type(8)))  float    v8f;

#define NSTATE      256
#define LSTEPS      256
#define ROWS_TOTAL  2048            // 16 batch * 128 ch
#define ROWS_PER_WG 16
#define KCHUNKS     8               // 256 / 32 (K per WMMA)
#define LDS_ROW     (NSTATE + 8)    // f16 elems per state row, +8 pad (528B) -> bank-conflict-free

// c_t = A c_{t-1} + f_t * B, streamed out every step.
// One WG = 16 sequence rows; 8 waves each own two 16-row n-tiles of A (A kept in VGPRs).
// State ping-pongs through padded LDS in f16; accumulate in f32 via v_wmma_f32_16x16x32_f16.
// The kc-loop is software-pipelined one stage (load b[kc+1] under wmma of b[kc]) and the
// schedule is pinned with sched_group_barrier so DS reads run ahead of the WMMA chain.
__global__ __launch_bounds__(256)
void lmu_scan_wmma(const float* __restrict__ inp,    // (2048, 256) = (row, t)
                   const float* __restrict__ Amat,   // (256, 256)
                   const float* __restrict__ Bvec,   // (256,)
                   float* __restrict__ out)          // (256, 2048, 256) = (t, row, n)
{
    __shared__ __align__(16) _Float16 Cst[2][ROWS_PER_WG * LDS_ROW];

    const int lane = threadIdx.x & 31;
    const int wave = threadIdx.x >> 5;     // 0..7 -> owns n-tiles 2w, 2w+1
    const int r    = lane & 15;            // row-in-chunk == N index of all tiles
    const int hi   = (lane >> 4) & 1;      // upper/lower lane half
    const int row0 = blockIdx.x * ROWS_PER_WG;

    // ---- c0 = 0 ----
    for (int i = threadIdx.x; i < ROWS_PER_WG * LDS_ROW; i += blockDim.x) {
        Cst[0][i] = (_Float16)0.f;
        Cst[1][i] = (_Float16)0.f;
    }

    // ---- A -> WMMA A-fragments, resident in VGPRs for all 256 steps ----
    // A-frag (16x32 f16), ISA 7.12.2: M = lane&15; halves 0..7 -> K = hi*8+e,
    // halves 8..15 -> K = 16 + hi*8 + e.
    v16h afrag[2][KCHUNKS];
    #pragma unroll
    for (int tt = 0; tt < 2; ++tt) {
        const int n0 = (2 * wave + tt) * 16;
        const float* arow = Amat + (size_t)(n0 + r) * NSTATE;
        #pragma unroll
        for (int kc = 0; kc < KCHUNKS; ++kc) {
            const int kLo = kc * 32 + hi * 8;
            v16h a;
            #pragma unroll
            for (int e = 0; e < 8; ++e) {
                a[e]     = (_Float16)arow[kLo + e];
                a[e + 8] = (_Float16)arow[kLo + 16 + e];
            }
            afrag[tt][kc] = a;
        }
    }

    // ---- per-lane slice of B matching the D-frag layout (VGPR v <-> n = base+v) ----
    float bv[2][8];
    #pragma unroll
    for (int tt = 0; tt < 2; ++tt) {
        const int nb = (2 * wave + tt) * 16 + hi * 8;
        #pragma unroll
        for (int v = 0; v < 8; ++v) bv[tt][v] = Bvec[nb + v];
    }

    const float* frow = inp + (size_t)(row0 + r) * LSTEPS;

    __syncthreads();

    int p = 0;
    for (int t = 0; t < LSTEPS; ++t) {
        // Issue the f_t load first so it overlaps the whole WMMA chain.
        const float ft = frow[t];

        // B-frag (32x16 f16): N = lane&15, half e<8 -> K = hi*8 + e, e>=8 -> K = 16 + hi*8 + (e-8)
        // => two 16-byte runs per lane in the padded [r][k] LDS image (two ds_load_b128).
        const _Float16* src = &Cst[p][r * LDS_ROW + hi * 8];

        v8f acc0 = {};
        v8f acc1 = {};

        // one-stage pipeline: bcur consumed by WMMA while bnext is loading
        v8h blo = *(const v8h*)(src);
        v8h bhi2 = *(const v8h*)(src + 16);
        v16h bcur;
        #pragma unroll
        for (int e = 0; e < 8; ++e) { bcur[e] = blo[e]; bcur[e + 8] = bhi2[e]; }

        #pragma unroll
        for (int kc = 0; kc < KCHUNKS; ++kc) {
            v16h bnext;
            if (kc + 1 < KCHUNKS) {
                v8h nlo = *(const v8h*)(src + (kc + 1) * 32);
                v8h nhi = *(const v8h*)(src + (kc + 1) * 32 + 16);
                #pragma unroll
                for (int e = 0; e < 8; ++e) { bnext[e] = nlo[e]; bnext[e + 8] = nhi[e]; }
            }
            acc0 = __builtin_amdgcn_wmma_f32_16x16x32_f16(false, afrag[0][kc],
                                                          false, bcur, (short)0, acc0,
                                                          false, false);
            acc1 = __builtin_amdgcn_wmma_f32_16x16x32_f16(false, afrag[1][kc],
                                                          false, bcur, (short)0, acc1,
                                                          false, false);
            if (kc + 1 < KCHUNKS) bcur = bnext;
        }

#if __has_builtin(__builtin_amdgcn_sched_group_barrier)
        // Pin the pipeline: ds(b0) ds(b1) wmma(b0) ds(b2) wmma(b1) ... wmma(b7)
        // 0x100 = DS read, 0x008 = MFMA/WMMA
        __builtin_amdgcn_sched_group_barrier(0x100, 2, 0);
        #pragma unroll
        for (int g = 0; g < KCHUNKS - 1; ++g) {
            __builtin_amdgcn_sched_group_barrier(0x100, 2, 0);
            __builtin_amdgcn_sched_group_barrier(0x008, 2, 0);
        }
        __builtin_amdgcn_sched_group_barrier(0x008, 2, 0);
#endif

        // inject f_t * B in f32
        #pragma unroll
        for (int v = 0; v < 8; ++v) {
            acc0[v] += ft * bv[0][v];
            acc1[v] += ft * bv[1][v];
        }

        // D-frag: VGPR v <-> n = tile*16 + hi*8 + v, at row r -> contiguous stores.
        _Float16* dstBase = &Cst[p ^ 1][r * LDS_ROW];
        float*    orow    = out + ((size_t)t * ROWS_TOTAL + row0 + r) * NSTATE;

        const int nb0 = (2 * wave + 0) * 16 + hi * 8;
        const int nb1 = (2 * wave + 1) * 16 + hi * 8;

        v8h h0, h1;
        #pragma unroll
        for (int v = 0; v < 8; ++v) {
            h0[v] = (_Float16)acc0[v];
            h1[v] = (_Float16)acc1[v];
        }
        *(v8h*)(dstBase + nb0) = h0;                     // next-step state (f16)
        *(v8h*)(dstBase + nb1) = h1;

        *(float4*)(orow + nb0)     = make_float4(acc0[0], acc0[1], acc0[2], acc0[3]);
        *(float4*)(orow + nb0 + 4) = make_float4(acc0[4], acc0[5], acc0[6], acc0[7]);
        *(float4*)(orow + nb1)     = make_float4(acc1[0], acc1[1], acc1[2], acc1[3]);
        *(float4*)(orow + nb1 + 4) = make_float4(acc1[4], acc1[5], acc1[6], acc1[7]);

        __syncthreads();
        p ^= 1;
    }
}

extern "C" void kernel_launch(void* const* d_in, const int* in_sizes, int n_in,
                              void* d_out, int out_size, void* d_ws, size_t ws_size,
                              hipStream_t stream) {
    (void)in_sizes; (void)n_in; (void)out_size; (void)d_ws; (void)ws_size;
    const float* inputs = (const float*)d_in[0];   // (16,128,256) f32
    const float* Amat   = (const float*)d_in[1];   // (256,256)    f32
    const float* Bvec   = (const float*)d_in[2];   // (256,)       f32
    float* out = (float*)d_out;                    // (256,16,128,256) f32

    dim3 grid(ROWS_TOTAL / ROWS_PER_WG);           // 128 workgroups
    dim3 block(256);                               // 8 wave32
    lmu_scan_wmma<<<grid, block, 0, stream>>>(inputs, Amat, Bvec, out);
}
